// FusedTransformerBlock_90941637525729
// MI455X (gfx1250) — matrix operand order repaired and checked by reading, not compiled, not run on hardware
//
#include <hip/hip_runtime.h>
#include <hip/hip_bf16.h>

typedef _Float16 f16;
typedef __attribute__((ext_vector_type(16))) _Float16 v16h;
typedef __attribute__((ext_vector_type(8)))  _Float16 v8h;
typedef __attribute__((ext_vector_type(4)))  _Float16 v4h;
typedef __attribute__((ext_vector_type(2)))  _Float16 v2h;
typedef __attribute__((ext_vector_type(8)))  float    v8f;

#define SEQ    2048
#define DMODEL 1024
#define NHEADS 16
#define HD     64
#define FFDIM  4096
#define MROWS  4096   // B*S

// ---------------------------------------------------------------------------
// CDNA5 async global->LDS copy (ASYNCcnt path).
// ---------------------------------------------------------------------------
#define USE_ASYNC_COPY 1

#if USE_ASYNC_COPY
__device__ __forceinline__ void async_ld_b128(const void* g, void* lds) {
  // VDST = LDS byte address (low 32 bits of flat pointer), VADDR = 64-bit global
  asm volatile("global_load_async_to_lds_b128 %0, %1, off"
               :: "v"((unsigned)(unsigned long long)(size_t)lds),
                  "v"((unsigned long long)(size_t)g)
               : "memory");
}
#define WAIT_ASYNC() asm volatile("s_wait_asynccnt 0x0" ::: "memory")
#else
__device__ __forceinline__ void async_ld_b128(const void* g, void* lds) {
  *(v8h*)lds = *(const v8h*)g;
}
#define WAIT_ASYNC()
#endif

// ---------------------------------------------------------------------------
// WMMA helpers (CDNA5 16x16x32 f16 -> f32)
// ---------------------------------------------------------------------------
__device__ __forceinline__ v8f wmma16(v16h a, v16h b, v8f c) {
  return __builtin_amdgcn_wmma_f32_16x16x32_f16(false, a, false, b, (short)0, c,
                                                false, false);
}

// A-fragment (16x32 f16): two contiguous 8-half chunks per lane.
__device__ __forceinline__ v16h make_frag(const f16* p0, const f16* p1) {
  v8h lo = *(const v8h*)p0;
  v8h hi = *(const v8h*)p1;
  return __builtin_shufflevector(lo, hi, 0, 1, 2, 3, 4, 5, 6, 7,
                                 8, 9, 10, 11, 12, 13, 14, 15);
}
__device__ __forceinline__ v16h cat16(v8h lo, v8h hi) {
  return __builtin_shufflevector(lo, hi, 0, 1, 2, 3, 4, 5, 6, 7,
                                 8, 9, 10, 11, 12, 13, 14, 15);
}
// B-fragment (32x16 f16): one contiguous 16-half chunk per lane.
__device__ __forceinline__ v16h load_b16(const f16* p) {
  return *(const v16h*)p;
}

// ---------------------------------------------------------------------------
// fp32 -> f16 conversion (4 elems/thread)
// ---------------------------------------------------------------------------
__global__ void cvt_f32_f16_kernel(const float* __restrict__ in,
                                   f16* __restrict__ out, int n) {
  int i = (blockIdx.x * 256 + threadIdx.x) * 4;
  if (i < n) {
    float4 f = *(const float4*)(in + i);
    v4h h;
    h.x = (f16)f.x; h.y = (f16)f.y; h.z = (f16)f.z; h.w = (f16)f.w;
    *(v4h*)(out + i) = h;
  }
}

// ---------------------------------------------------------------------------
// GEMM: C[M,N] = A[M,K](f16) * B[K,N](f16) + bias.
// 128x128 block tile, 8 waves, BK=32, LDS double-buffered; A-tile via async
// global->LDS, B-tile transposed in LDS with packed b32 stores.
// EPI: 0=f32 out, 1=f16 out, 2=f16 out+GELU(erf), 3=f16 out scaled by 0.125
// ---------------------------------------------------------------------------
template <int EPI>
__global__ __launch_bounds__(256)
void gemm_f16_kernel(const f16* __restrict__ A, const f16* __restrict__ B,
                     const float* __restrict__ bias, float* __restrict__ outF,
                     f16* __restrict__ outH, int M, int N, int K) {
  __shared__ f16 At[2][128][32];   // [buf][m][k]
  __shared__ f16 Bt[2][128][32];   // [buf][n][k] (transposed)

  const int tid  = threadIdx.x;
  const int lane = tid & 31;
  const int wid  = tid >> 5;
  const int half = lane >> 4;
  const int ln16 = lane & 15;
  const int kbA  = half ? 8 : 0;
  const int kbB  = half ? 16 : 0;

  const int n0 = blockIdx.x * 128;
  const int m0 = blockIdx.y * 128;
  const int wm = (wid & 3) * 32;
  const int wn = (wid >> 2) * 64;

  const int lam = tid >> 1;
  const int lak = (tid & 1) * 16;
  const int kpB = (tid & 15) * 2;
  const int nsB = (tid >> 4) * 8;

  const f16* Abase = A + (size_t)(m0 + lam) * K + lak;
  const f16* Bbase = B + n0 + nsB;

  v8f acc[2][4] = {};
  v8h br0, br1;

  auto loadA = [&](int k0, int buf) {
    async_ld_b128(Abase + k0,     &At[buf][lam][lak]);
    async_ld_b128(Abase + k0 + 8, &At[buf][lam][lak + 8]);
  };
  auto loadB = [&](int k0) {
    const f16* p = Bbase + (size_t)(k0 + kpB) * N;
    br0 = *(const v8h*)p;
    br1 = *(const v8h*)(p + N);
  };
  auto storeB = [&](int buf) {
#pragma unroll
    for (int i = 0; i < 8; ++i) {
      v2h pk; pk.x = br0[i]; pk.y = br1[i];
      *(v2h*)&Bt[buf][nsB + i][kpB] = pk;
    }
  };

  loadA(0, 0);
  loadB(0);
  storeB(0);
  WAIT_ASYNC();
  __syncthreads();

  const int nk = K / 32;
  for (int kt = 0; kt < nk; ++kt) {
    const int cur  = kt & 1;
    const bool more = (kt + 1) < nk;
    if (more) {
      loadA((kt + 1) * 32, cur ^ 1);
      loadB((kt + 1) * 32);
    }

    v16h af[2], bf[4];
#pragma unroll
    for (int i = 0; i < 2; ++i)
      af[i] = make_frag(&At[cur][wm + i * 16 + ln16][kbB],
                        &At[cur][wm + i * 16 + ln16][kbB + 8]);
#pragma unroll
    for (int j = 0; j < 4; ++j)
      bf[j] = load_b16(&Bt[cur][wn + j * 16 + ln16][kbB]);
#pragma unroll
    for (int i = 0; i < 2; ++i)
#pragma unroll
      for (int j = 0; j < 4; ++j)
        acc[i][j] = wmma16(af[i], bf[j], acc[i][j]);

    if (more) {
      storeB(cur ^ 1);
      WAIT_ASYNC();
    }
    __syncthreads();
  }

#pragma unroll
  for (int i = 0; i < 2; ++i)
#pragma unroll
    for (int j = 0; j < 4; ++j)
#pragma unroll
      for (int r = 0; r < 8; ++r) {
        int row = m0 + wm + i * 16 + r + half * 8;
        int col = n0 + wn + j * 16 + ln16;
        float v = acc[i][j][r] + bias[col];
        if (EPI == 2) v = 0.5f * v * (1.0f + erff(v * 0.70710678118654752f));
        if (EPI == 3) v *= 0.125f;   // 1/sqrt(HD) folded into Q
        if (EPI == 0)
          outF[(size_t)row * N + col] = v;
        else
          outH[(size_t)row * N + col] = (f16)v;
      }
}

// ---------------------------------------------------------------------------
// Flash attention, transposed formulation. Per wave: 16 q rows.
//   S^T = K * Q^T      (K rows = A-frags from async-staged LDS, Q^T = B-frags)
//   P^T = exp(S^T)     (C-layout has q fixed per lane -> B-frag needs only an
//                       8-half partner swap via __shfl_xor(...,16))
//   ctx^T = V^T * P^T  (V^T A-frags from transposed LDS tile)
//   l     = ones * P^T (column sums broadcast; one 1/l per lane)
// Q is pre-scaled by 1/8; softmax max-subtraction skipped (scores are O(1)).
// ---------------------------------------------------------------------------
__global__ __launch_bounds__(128)
void attn_kernel(const f16* __restrict__ Qh, const f16* __restrict__ Kh,
                 const f16* __restrict__ Vh, f16* __restrict__ ctxh) {
  __shared__ f16 Ktile[32][64];   // [key][d]
  __shared__ f16 Vt[64][32];      // [d][key] (transposed)

  const int tid   = threadIdx.x;
  const int lane  = tid & 31;
  const int wid   = tid >> 5;
  const int half  = lane >> 4;
  const int ln16  = lane & 15;
  const int kbA   = half ? 8 : 0;
  const int kbB   = half ? 16 : 0;

  const int qb    = blockIdx.x;
  const int bh    = blockIdx.y;
  const int b     = bh >> 4;
  const int h     = bh & 15;
  const int hbase = h * HD;
  const int qrow0 = b * SEQ + qb * 64 + wid * 16;

  // Q^T B-fragments (persist across the whole loop): lane = column q
  v16h qb0, qb1;
  {
    const f16* qr = Qh + (size_t)(qrow0 + ln16) * DMODEL + hbase;
    qb0 = load_b16(qr + kbB);        // d 0..31
    qb1 = load_b16(qr + 32 + kbB);   // d 32..63
  }

  v16h onesA;
#pragma unroll
  for (int i = 0; i < 16; ++i) onesA[i] = (_Float16)1.0f;

  v8f acc[4] = {};   // ctx^T d-tiles
  v8f lacc = {};     // column sums

  const int kk_ld = tid & 31;
  const int dseg  = tid >> 5;
  const int kpV = (tid & 15) * 2;
  const int dsV = (tid >> 4) * 8;

  for (int kc = 0; kc < SEQ / 32; ++kc) {
    __syncthreads();
    const int krow = b * SEQ + kc * 32;
    {
      const f16* kb = Kh + (size_t)(krow + kk_ld) * DMODEL + hbase + dseg * 16;
      async_ld_b128(kb,     &Ktile[kk_ld][dseg * 16]);
      async_ld_b128(kb + 8, &Ktile[kk_ld][dseg * 16 + 8]);
      const f16* vb = Vh + (size_t)(krow + kpV) * DMODEL + hbase + dsV;
      v8h v0 = *(const v8h*)vb;
      v8h v1 = *(const v8h*)(vb + DMODEL);
#pragma unroll
      for (int i = 0; i < 8; ++i) {
        v2h pk; pk.x = v0[i]; pk.y = v1[i];
        *(v2h*)&Vt[dsV + i][kpV] = pk;
      }
      WAIT_ASYNC();
    }
    __syncthreads();

    // S^T tiles: st0 = kk 0..15, st1 = kk 16..31 (rows), q = lane column
    v8f st0 = {}, st1 = {};
    {
      const f16* k0r = &Ktile[ln16][0];
      const f16* k1r = &Ktile[16 + ln16][0];
      st0 = wmma16(make_frag(k0r + kbB, k0r + 8 + kbB), qb0, st0);
      st0 = wmma16(make_frag(k0r + 32 + kbB, k0r + 40 + kbB), qb1, st0);
      st1 = wmma16(make_frag(k1r + kbB, k1r + 8 + kbB), qb0, st1);
      st1 = wmma16(make_frag(k1r + 32 + kbB, k1r + 40 + kbB), qb1, st1);
    }

    // P^T = exp(S^T), packed to f16
    union { v8h h; unsigned u[4]; } p0, p1, e0, e1;
#pragma unroll
    for (int r = 0; r < 8; ++r) {
      p0.h[r] = (f16)__expf(st0[r]);
      p1.h[r] = (f16)__expf(st1[r]);
    }
    // partner-half swap: build B-fragment of P^T (32kk x 16q)
#pragma unroll
    for (int i = 0; i < 4; ++i) {
      e0.u[i] = __shfl_xor(p0.u[i], 16, 32);
      e1.u[i] = __shfl_xor(p1.u[i], 16, 32);
    }
    v8h lo = half ? e1.h : p0.h;   // kk kbB..kbB+7
    v8h hi = half ? p1.h : e0.h;   // kk kbB+8..kbB+15
    v16h pb = cat16(lo, hi);

    lacc = wmma16(onesA, pb, lacc);              // column sums (broadcast)
#pragma unroll
    for (int t = 0; t < 4; ++t) {
      const f16* vr = &Vt[t * 16 + ln16][0];
      acc[t] = wmma16(make_frag(vr + kbB, vr + 8 + kbB), pb, acc[t]);
    }
  }

  // ctx^T -> ctx: lane owns row q = qrow0+ln16; consecutive r = consecutive d
  {
    float inv = 1.0f / lacc[0];
    f16* orow = ctxh + (size_t)(qrow0 + ln16) * DMODEL + hbase;
#pragma unroll
    for (int t = 0; t < 4; ++t) {
      v8h o;
#pragma unroll
      for (int r = 0; r < 8; ++r) o[r] = (f16)(acc[t][r] * inv);
      *(v8h*)(orow + t * 16 + half * 8) = o;
    }
  }
}

// ---------------------------------------------------------------------------
// LayerNorm(a + c) * g + beta ; f32 out (+ optional f16 copy).
// ---------------------------------------------------------------------------
__global__ __launch_bounds__(256)
void ln_residual_kernel(const float* __restrict__ a, const float* __restrict__ c,
                        const float* __restrict__ g, const float* __restrict__ beta,
                        float* __restrict__ out32, f16* __restrict__ out16) {
  const int row = blockIdx.x;
  const int tid = threadIdx.x;
  __shared__ float red[256];
  const float* pa = a + (size_t)row * DMODEL;
  const float* pc = c + (size_t)row * DMODEL;

  float v[4];
  float s = 0.0f;
#pragma unroll
  for (int i = 0; i < 4; ++i) {
    int idx = tid + i * 256;
    float t = pa[idx] + pc[idx];
    v[i] = t;
    s += t;
  }
  red[tid] = s;
  __syncthreads();
  for (int off = 128; off > 0; off >>= 1) {
    if (tid < off) red[tid] += red[tid + off];
    __syncthreads();
  }
  float mu = red[0] * (1.0f / (float)DMODEL);
  __syncthreads();

  float q = 0.0f;
#pragma unroll
  for (int i = 0; i < 4; ++i) { float d = v[i] - mu; q += d * d; }
  red[tid] = q;
  __syncthreads();
  for (int off = 128; off > 0; off >>= 1) {
    if (tid < off) red[tid] += red[tid + off];
    __syncthreads();
  }
  float rstd = rsqrtf(red[0] * (1.0f / (float)DMODEL) + 1e-5f);

#pragma unroll
  for (int i = 0; i < 4; ++i) {
    int idx = tid + i * 256;
    float y = (v[i] - mu) * rstd * g[idx] + beta[idx];
    out32[(size_t)row * DMODEL + idx] = y;
    if (out16) out16[(size_t)row * DMODEL + idx] = (f16)y;
  }
}

// ---------------------------------------------------------------------------
// Host orchestration
// ---------------------------------------------------------------------------
extern "C" void kernel_launch(void* const* d_in, const int* in_sizes, int n_in,
                              void* d_out, int out_size, void* d_ws, size_t ws_size,
                              hipStream_t stream) {
  (void)in_sizes; (void)n_in; (void)out_size; (void)ws_size;

  const float* x  = (const float*)d_in[0];
  const float* Wq = (const float*)d_in[1];
  const float* bq = (const float*)d_in[2];
  const float* Wk = (const float*)d_in[3];
  const float* bk = (const float*)d_in[4];
  const float* Wv = (const float*)d_in[5];
  const float* bv = (const float*)d_in[6];
  const float* Wo = (const float*)d_in[7];
  const float* bo = (const float*)d_in[8];
  const float* W1 = (const float*)d_in[9];
  const float* b1 = (const float*)d_in[10];
  const float* W2 = (const float*)d_in[11];
  const float* b2 = (const float*)d_in[12];
  const float* g1  = (const float*)d_in[13];
  const float* be1 = (const float*)d_in[14];
  const float* g2  = (const float*)d_in[15];
  const float* be2 = (const float*)d_in[16];
  float* out = (float*)d_out;

  char* ws = (char*)d_ws;
  const size_t MB = 1024ull * 1024ull;
  f16*   xh    = (f16*)(ws + 0);          //  8 MB
  f16*   Qh    = (f16*)(ws + 8 * MB);     //  8 MB (pre-scaled by 1/8)
  f16*   Kh    = (f16*)(ws + 16 * MB);    //  8 MB
  f16*   Vh    = (f16*)(ws + 24 * MB);    //  8 MB
  f16*   ctxh  = (f16*)(ws + 32 * MB);    //  8 MB
  f16*   hh    = (f16*)(ws + 40 * MB);    //  8 MB
  f16*   ffh   = (f16*)(ws + 48 * MB);    // 32 MB
  f16*   Wqh   = (f16*)(ws + 80 * MB);    //  2 MB
  f16*   Wkh   = (f16*)(ws + 82 * MB);    //  2 MB
  f16*   Wvh   = (f16*)(ws + 84 * MB);    //  2 MB
  f16*   Woh   = (f16*)(ws + 86 * MB);    //  2 MB
  f16*   W1h   = (f16*)(ws + 88 * MB);    //  8 MB
  f16*   W2h   = (f16*)(ws + 96 * MB);    //  8 MB
  float* attnF = (float*)(ws + 104 * MB); // 16 MB
  float* hbuf  = (float*)(ws + 120 * MB); // 16 MB
  float* ff2F  = (float*)(ws + 136 * MB); // 16 MB

  auto cvt = [&](const float* src, f16* dst, int n) {
    cvt_f32_f16_kernel<<<(n / 4 + 255) / 256, 256, 0, stream>>>(src, dst, n);
  };

  cvt(x,  xh,  MROWS * DMODEL);
  cvt(Wq, Wqh, DMODEL * DMODEL);
  cvt(Wk, Wkh, DMODEL * DMODEL);
  cvt(Wv, Wvh, DMODEL * DMODEL);
  cvt(Wo, Woh, DMODEL * DMODEL);
  cvt(W1, W1h, DMODEL * FFDIM);
  cvt(W2, W2h, FFDIM * DMODEL);

  dim3 gQKV(DMODEL / 128, MROWS / 128);
  gemm_f16_kernel<3><<<gQKV, 256, 0, stream>>>(xh, Wqh, bq, nullptr, Qh,
                                               MROWS, DMODEL, DMODEL);
  gemm_f16_kernel<1><<<gQKV, 256, 0, stream>>>(xh, Wkh, bk, nullptr, Kh,
                                               MROWS, DMODEL, DMODEL);
  gemm_f16_kernel<1><<<gQKV, 256, 0, stream>>>(xh, Wvh, bv, nullptr, Vh,
                                               MROWS, DMODEL, DMODEL);

  attn_kernel<<<dim3(SEQ / 64, 2 * NHEADS), 128, 0, stream>>>(Qh, Kh, Vh, ctxh);

  gemm_f16_kernel<0><<<gQKV, 256, 0, stream>>>(ctxh, Woh, bo, attnF, nullptr,
                                               MROWS, DMODEL, DMODEL);

  ln_residual_kernel<<<MROWS, 256, 0, stream>>>(x, attnF, g1, be1, hbuf, hh);

  dim3 gFF1(FFDIM / 128, MROWS / 128);
  gemm_f16_kernel<2><<<gFF1, 256, 0, stream>>>(hh, W1h, b1, nullptr, ffh,
                                               MROWS, FFDIM, DMODEL);

  gemm_f16_kernel<0><<<gQKV, 256, 0, stream>>>(ffh, W2h, b2, ff2F, nullptr,
                                               MROWS, DMODEL, FFDIM);

  ln_residual_kernel<<<MROWS, 256, 0, stream>>>(hbuf, ff2F, g2, be2, out, nullptr);
}
